// HACCutLayer_1906965479891
// MI455X (gfx1250) — 8-bit matrix operand order repaired and checked by reading, compiled, not run on hardware
//
#include <hip/hip_runtime.h>
#include <math.h>

typedef int v8i __attribute__((ext_vector_type(8)));

#define HAC_D 1024

__device__ __forceinline__ void waveArgmin(float& v, int& i) {
#pragma unroll
  for (int off = 16; off > 0; off >>= 1) {
    float ov = __shfl_down(v, off, 32);
    int   oi = __shfl_down(i, off, 32);
    if (ov < v || (ov == v && oi < i)) { v = ov; i = oi; }
  }
}
__device__ __forceinline__ float waveMax(float v) {
#pragma unroll
  for (int off = 16; off > 0; off >>= 1) v = fmaxf(v, __shfl_down(v, off, 32));
  return v;
}

__global__ void cr_init_kernel(const float* __restrict__ Wm, float* __restrict__ CR, int D) {
  int idx = blockIdx.x * blockDim.x + threadIdx.x;
  if (idx < D * D) {
    int i = idx / D, j = idx - i * D;
    CR[idx] = Wm[idx] + Wm[j * D + i];
  }
}

__global__ __launch_bounds__(1024, 1)
void hac_seq_kernel(const float* __restrict__ Xg, float* __restrict__ Yg,
                    float* __restrict__ CRg, int* __restrict__ labg, int D) {
  __shared__ float values[HAC_D];
  __shared__ int   indices[HAC_D];
  __shared__ float newv[HAC_D];
  __shared__ float vtmp[HAC_D];
  __shared__ float cs_l[HAC_D];
  __shared__ float S_l[HAC_D];
  __shared__ float E_l[HAC_D];
  __shared__ int   member_l[HAC_D];
  __shared__ int   label_l[HAC_D];
  __shared__ int   alive_l[HAC_D];
  __shared__ int   resc[HAC_D];
  __shared__ float redv[32];
  __shared__ int   redi[32];
  __shared__ float redv2[32];
  __shared__ int   sm1, sm2, snresc, stake;
  __shared__ float scs1, scs2, sncs, sMAXD, sMS;

  const int tid  = threadIdx.x;
  const int lane = tid & 31;
  const int wid  = tid >> 5;
  const int NT   = blockDim.x;

  float la = 0.0f, lm = -3.4e38f;
  for (int i = tid; i < D * D; i += NT) {
    float x = Xg[i];
    la = fmaxf(la, fabsf(x));
    lm = fmaxf(lm, x);
  }
  la = waveMax(la); lm = waveMax(lm);
  if (lane == 0) { redv[wid] = la; redv2[wid] = lm; }
  __syncthreads();
  if (wid == 0) {
    la = waveMax(redv[lane]); lm = waveMax(redv2[lane]);
    if (lane == 0) { sMAXD = la * 1000.0f; sMS = lm; }
  }
  __syncthreads();
  const float MAXD = sMAXD, MS = sMS;

  for (int t = tid; t < D; t += NT) {
    float mn = MAXD; int mi = 0;
    const float* xr = Xg + (size_t)t * D;
    float* yr = Yg + (size_t)t * D;
    for (int j = 0; j < D; ++j) {
      float v = (j > t) ? (MS - xr[j]) : MAXD;
      yr[j] = v;
      if (v < mn) { mn = v; mi = j; }
    }
    values[t] = mn; indices[t] = mi;
    cs_l[t] = 1.0f; S_l[t] = 0.0f; E_l[t] = 0.0f;
    member_l[t] = t; label_l[t] = t; alive_l[t] = 1;
  }
  __syncthreads();

  for (int it = 0; it < D - 1; ++it) {
    float v = values[tid];
    int ix = tid;
    waveArgmin(v, ix);
    if (lane == 0) { redv[wid] = v; redi[wid] = ix; }
    __syncthreads();
    if (wid == 0) {
      v = redv[lane]; ix = redi[lane];
      waveArgmin(v, ix);
      if (lane == 0) {
        int m1 = ix;
        int m2 = indices[m1];
        sm1 = m1; sm2 = m2;
        float c1 = cs_l[m1], c2 = cs_l[m2];
        scs1 = c1; scs2 = c2; sncs = c1 + c2;
        cs_l[m1] = c1 + c2;
        float cross = CRg[(size_t)m1 * D + m2];
        float newS  = S_l[m1] + S_l[m2] + cross;
        float esum  = E_l[m1] + E_l[m2];
        int take = (newS >= esum) ? 1 : 0;
        stake = take;
        S_l[m1] = newS;
        E_l[m1] = take ? newS : esum;
        alive_l[m2] = 0;
        values[m2] = MAXD;
        snresc = 0;
      }
    }
    __syncthreads();
    const int m1 = sm1, m2 = sm2, take = stake;
    const float cs1 = scs1, cs2 = scs2, ncs = sncs;

    for (int j = tid; j < D; j += NT) {
      float a = (j > m1) ? Yg[(size_t)m1 * D + j] : ((j < m1) ? Yg[(size_t)j * D + m1] : MAXD);
      float b = (j > m2) ? Yg[(size_t)m2 * D + j] : ((j < m2) ? Yg[(size_t)j * D + m2] : MAXD);
      newv[j] = (a * cs1 + b * cs2) / ncs;
      vtmp[j] = CRg[(size_t)m1 * D + j] + CRg[(size_t)m2 * D + j];
      int mm = member_l[j];
      if (mm == m2) { mm = m1; member_l[j] = m1; }
      if (take && mm == m1) label_l[j] = m1;
    }
    __syncthreads();

    for (int j = tid; j < D; j += NT) {
      float cv = vtmp[j];
      CRg[(size_t)m1 * D + j] = cv;
      CRg[(size_t)j * D + m1] = cv;
      float nv = newv[j];
      int aj = alive_l[j];
      Yg[(size_t)m1 * D + j] = (j > m1 && aj) ? nv : MAXD;
      Yg[(size_t)j * D + m1] = (j < m1 && aj) ? nv : MAXD;
      Yg[(size_t)m2 * D + j] = MAXD;
      Yg[(size_t)j * D + m2] = MAXD;
    }
    __syncthreads();

    for (int j = tid; j < D; j += NT) {
      if (j != m1 && alive_l[j] && values[j] != MAXD) {
        int oi = indices[j];
        if (oi == m1 || oi == m2) {
          int p = atomicAdd(&snresc, 1);
          resc[p] = j;
        } else if (j < m1) {
          float c = newv[j];
          if (c < values[j]) { values[j] = c; indices[j] = m1; }
        }
      }
    }
    __syncthreads();

    float c = (tid > m1 && alive_l[tid]) ? newv[tid] : MAXD;
    int ci = tid;
    waveArgmin(c, ci);
    if (lane == 0) { redv[wid] = c; redi[wid] = ci; }
    __syncthreads();
    if (wid == 0) {
      c = redv[lane]; ci = redi[lane];
      waveArgmin(c, ci);
      if (lane == 0) { values[m1] = c; indices[m1] = (c == MAXD) ? 0 : ci; }
    }

    int nr = snresc;
    for (int ii = wid; ii < nr; ii += (NT >> 5)) {
      int r = resc[ii];
      float bv = MAXD; int bi = 0;
      const float* row = Yg + (size_t)r * D;
      for (int k = lane; k < D; k += 32) {
        float vv = row[k];
        if (vv < bv || (vv == bv && k < bi)) { bv = vv; bi = k; }
      }
      waveArgmin(bv, bi);
      if (lane == 0) { values[r] = bv; indices[r] = (bv == MAXD) ? 0 : bi; }
    }
    __syncthreads();
  }

  for (int t = tid; t < D; t += NT) labg[t] = label_l[t];
}

__global__ void u_build_kernel(const int* __restrict__ lab, signed char* __restrict__ U, int D) {
  int idx = blockIdx.x * blockDim.x + threadIdx.x;
  if (idx < D * D) {
    int x = idx / D, c = idx - x * D;
    U[idx] = (lab[x] == c) ? (signed char)1 : (signed char)0;
  }
}

__global__ __launch_bounds__(256, 1)
void uut_wmma_kernel(const signed char* __restrict__ U, float* __restrict__ R, int D) {
  const int gwave  = (int)((blockIdx.x * blockDim.x + threadIdx.x) >> 5);
  const int lane   = threadIdx.x & 31;
  const int nTileN = D >> 4;
  const int nTiles = nTileN * nTileN;
  const int totW   = (int)((gridDim.x * blockDim.x) >> 5);
  const int hi     = lane >> 4;
  const int l15    = lane & 15;

  for (int t = gwave; t < nTiles; t += totW) {
    const int tm = (t / nTileN) << 4;
    const int tn = (t - (t / nTileN) * nTileN) << 4;
    const signed char* arow = U + (size_t)(tm + l15) * D;
    const signed char* brow = U + (size_t)(tn + l15) * D;
    v8i acc = {};
    for (int k0 = 0; k0 < D; k0 += 64) {
      const int* ap = (const int*)(arow + k0 + hi * 8);
      const int* bp = (const int*)(brow + k0 + hi * 8);
      v8i a, b;
      a[0] = ap[0];  a[1] = ap[1];  a[2] = ap[4];  a[3] = ap[5];
      a[4] = ap[8];  a[5] = ap[9];  a[6] = ap[12]; a[7] = ap[13];
      b[0] = bp[0];  b[1] = bp[1];  b[2] = bp[4];  b[3] = bp[5];
      b[4] = bp[8];  b[5] = bp[9];  b[6] = bp[12]; b[7] = bp[13];
      acc = __builtin_amdgcn_wmma_i32_16x16x64_iu8(false, a, false, b, acc, false, false);
    }
    const int col   = tn + l15;
    const int rbase = tm + hi * 8;
#pragma unroll
    for (int vv = 0; vv < 8; ++vv) {
      R[(size_t)(rbase + vv) * D + col] = (float)acc[vv];
    }
  }
}

extern "C" void kernel_launch(void* const* d_in, const int* in_sizes, int n_in,
                              void* d_out, int out_size, void* d_ws, size_t ws_size,
                              hipStream_t stream) {
  (void)n_in; (void)out_size;
  const float* X  = (const float*)d_in[0];
  const float* Wm = (const float*)d_in[1];

  int D = 1;
  while ((long long)D * D < (long long)in_sizes[0]) ++D;

  const size_t mat = (size_t)D * D;
  float* Y  = (float*)d_out;
  float* CR = (float*)d_ws;
  int*   lab = (int*)((char*)d_ws + mat * sizeof(float));
  signed char* U = (signed char*)d_ws;

  if (ws_size < mat * sizeof(float) + (size_t)D * sizeof(int)) return;

  const int tpb = 256;
  const int gridE = (int)((mat + tpb - 1) / tpb);

  cr_init_kernel<<<gridE, tpb, 0, stream>>>(Wm, CR, D);
  hac_seq_kernel<<<1, 1024, 0, stream>>>(X, Y, CR, lab, D);
  u_build_kernel<<<gridE, tpb, 0, stream>>>(lab, U, D);

  const int tiles = (D / 16) * (D / 16);
  const int wavesPerBlock = tpb / 32;
  const int blocks = (tiles + wavesPerBlock - 1) / wavesPerBlock;
  uut_wmma_kernel<<<blocks, tpb, 0, stream>>>(U, (float*)d_out, D);
}
